// SSMB_1735166788054
// MI455X (gfx1250) — matrix operand order repaired and checked by reading, not compiled, not run on hardware
//
#include <hip/hip_runtime.h>
#include <hip/hip_bf16.h>

// ---------------------------------------------------------------------------
// Mamba block for MI455X (gfx1250, wave32, WMMA).
//  - GEMMs: v_wmma_f32_16x16x32_bf16, f32->bf16 inline at fragment load,
//    double-buffered raw-fragment registers so loads overlap WMMA; tail
//    peeled so the steady-state inner loop is branch-free.
//  - Scan: 3-phase chunked linear recurrence (16 chunks of 128) for 16x
//    parallelism over the sequential dimension.
// ---------------------------------------------------------------------------

typedef __attribute__((ext_vector_type(16))) __bf16 bf16x16;
typedef __attribute__((ext_vector_type(8)))  float  f32x8;

constexpr int kB   = 2;
constexpr int kL   = 2048;
constexpr int kDM  = 1024;           // d_model
constexpr int kDH  = 1024;           // d_inner/2
constexpr int kDI  = 2048;           // d_inner
constexpr int kRNK = 64;             // dt_rank
constexpr int kNS  = 16;             // d_state
constexpr int kXD  = kRNK + 2*kNS;   // 96
constexpr int kM   = kB * kL;        // 4096 rows for all GEMMs
constexpr int kNC  = 16;             // scan chunks
constexpr int kCL  = kL / kNC;       // 128 steps per chunk

// Workspace layout (floats). Total ~26.6M floats (~107 MB) — L2-resident.
constexpr size_t OFF_XZ    = 0;                                   // (kM,kDI)
constexpr size_t OFF_U     = OFF_XZ    + (size_t)kM * kDI;        // (kM,kDH)
constexpr size_t OFF_XDBL  = OFF_U     + (size_t)kM * kDH;        // (kM,kXD)
constexpr size_t OFF_DELTA = OFF_XDBL  + (size_t)kM * kXD;        // (kM,kDH)
constexpr size_t OFF_YCAT  = OFF_DELTA + (size_t)kM * kDH;        // (kM,kDI)
constexpr size_t OFF_SEND  = OFF_YCAT  + (size_t)kM * kDI;        // (B,NC,NS,DH)
constexpr size_t OFF_SINIT = OFF_SEND  + (size_t)kB*kNC*kNS*kDH;  // (B,NC,NS,DH)
constexpr size_t OFF_SUMDT = OFF_SINIT + (size_t)kB*kNC*kNS*kDH;  // (B,NC,DH)

__device__ __forceinline__ float siluf(float x) {
  return x * (1.0f / (1.0f + __expf(-x)));
}
__device__ __forceinline__ float softplusf(float x) {
  return (x > 20.0f) ? x : log1pf(__expf(x));
}

// Pack two float4 (8 contiguous f32) into bf16x16[BASE..BASE+7].
template <int BASE>
__device__ __forceinline__ void put8(bf16x16& v, float4 a, float4 b) {
  v[BASE + 0] = (__bf16)a.x; v[BASE + 1] = (__bf16)a.y;
  v[BASE + 2] = (__bf16)a.z; v[BASE + 3] = (__bf16)a.w;
  v[BASE + 4] = (__bf16)b.x; v[BASE + 5] = (__bf16)b.y;
  v[BASE + 6] = (__bf16)b.z; v[BASE + 7] = (__bf16)b.w;
}

// ---------------------------------------------------------------------------
// WMMA GEMM: C(M,N) = A(M,K) * W(N,K)^T    (row-major, K contiguous)
// Wave computes a 16x(16*NB) strip. Tile index is m-fastest so concurrent
// waves share W tiles (L0/L2 broadcast). Inner loop is double-buffered with
// a peeled tail: the hot body is fetch/compute/fetch/compute, branch-free.
// A striping (16-bit 16x32): lane<16 row m K{0..15}; lane>=16 row m
// K{16..31}. B: lane n / n+16 hold col n K{0..15}/{16..31}.
// C: VGPR r -> M=r (lanes 0-15) / M=8+r (lanes 16-31), N=lane%16.
// EPI==1: fused delta = softplus(acc + 2*bias[col]) (dt_proj).
// ---------------------------------------------------------------------------
template <int NB, int EPI>
__launch_bounds__(256)
__global__ void wmma_gemm_f32bf16(const float* __restrict__ A,
                                  const float* __restrict__ W,
                                  float* __restrict__ C,
                                  const float* __restrict__ bias,
                                  int M, int N, int K,
                                  int lda, int ldw, int ldc) {
  const int lane    = threadIdx.x & 31;
  const int wave    = blockIdx.x * (blockDim.x >> 5) + (threadIdx.x >> 5);
  const int mtiles  = M >> 4;
  const int ngroups = N / (16 * NB);
  if (wave >= mtiles * ngroups) return;       // wave-uniform: EXEC stays all-1s

  const int mi   = wave % mtiles;             // m-fastest: waves share W tiles
  const int ni   = wave / mtiles;
  const int m0   = mi << 4;
  const int n0   = ni * (16 * NB);
  const int half = lane >> 4;
  const int l16  = lane & 15;

  f32x8 acc[NB];
#pragma unroll
  for (int i = 0; i < NB; ++i) acc[i] = (f32x8)0.0f;

  const float* aptr = A + (size_t)(m0 + l16) * lda + half * 16;
  const float* wptr[NB];
#pragma unroll
  for (int nb = 0; nb < NB; ++nb)
    wptr[nb] = W + (size_t)(n0 + nb * 16 + l16) * ldw + half * 16;

  float4 ra[2][4];
  float4 rb[2][NB][4];

  auto fetch = [&](int k0, float4 (&fa)[4], float4 (&fb)[NB][4]) {
    fa[0] = *reinterpret_cast<const float4*>(aptr + k0);
    fa[1] = *reinterpret_cast<const float4*>(aptr + k0 + 4);
    fa[2] = *reinterpret_cast<const float4*>(aptr + k0 + 8);
    fa[3] = *reinterpret_cast<const float4*>(aptr + k0 + 12);
#pragma unroll
    for (int nb = 0; nb < NB; ++nb) {
#pragma unroll
      for (int j = 0; j < 4; ++j)
        fb[nb][j] = *reinterpret_cast<const float4*>(wptr[nb] + k0 + j * 4);
    }
  };

  auto compute = [&](float4 (&fa)[4], float4 (&fb)[NB][4]) {
    bf16x16 af;
    put8<0>(af, fa[0], fa[1]);
    put8<8>(af, fa[2], fa[3]);
#pragma unroll
    for (int nb = 0; nb < NB; ++nb) {
      bf16x16 bf;
      put8<0>(bf, fb[nb][0], fb[nb][1]);
      put8<8>(bf, fb[nb][2], fb[nb][3]);
      acc[nb] = __builtin_amdgcn_wmma_f32_16x16x32_bf16(
          false, af, false, bf, (short)0, acc[nb], false, false);
    }
  };

  // K is always a multiple of 64 here (1024 / 2048 / 64).
  fetch(0, ra[0], rb[0]);
  int k0 = 0;
  for (; k0 + 64 < K; k0 += 64) {     // steady state: branch-free body
    fetch(k0 + 32, ra[1], rb[1]);
    compute(ra[0], rb[0]);
    fetch(k0 + 64, ra[0], rb[0]);
    compute(ra[1], rb[1]);
  }
  // tail: k0 == K - 64
  fetch(k0 + 32, ra[1], rb[1]);
  compute(ra[0], rb[0]);
  compute(ra[1], rb[1]);

  // ---- store C (optionally fused softplus epilogue) ----
#pragma unroll
  for (int nb = 0; nb < NB; ++nb) {
    const int col = n0 + nb * 16 + l16;
    const float badd = (EPI == 1) ? 2.0f * bias[col] : 0.0f;
#pragma unroll
    for (int r = 0; r < 8; ++r) {
      const int row = m0 + r + half * 8;
      float v = acc[nb][r];
      if (EPI == 1) v = softplusf(v + badd);
      C[(size_t)row * ldc + col] = v;
    }
  }
}

// ---------------------------------------------------------------------------
// Depthwise conv (K=4, pad 1 left / 2 right) + SiLU for both halves of xz.
// x-half -> u (B,L,DH); z-half -> ycat columns [1024,2048).
// ---------------------------------------------------------------------------
__launch_bounds__(256)
__global__ void conv_silu_kernel(const float* __restrict__ xz,
                                 const float* __restrict__ wx,
                                 const float* __restrict__ wz,
                                 float* __restrict__ u,
                                 float* __restrict__ ycat) {
  const int idx = blockIdx.x * blockDim.x + threadIdx.x;  // over kM*kDI
  const int d2 = idx & (kDI - 1);
  const int bl = idx >> 11;           // kDI == 2^11
  const int l  = bl & (kL - 1);
  const int b  = bl >> 11;            // kL == 2^11

  const float* w = (d2 < kDH) ? (wx + d2 * 4) : (wz + (d2 - kDH) * 4);
  float acc = 0.0f;
#pragma unroll
  for (int k = 0; k < 4; ++k) {
    const int ll = l - 1 + k;         // pad_left = 1
    if (ll >= 0 && ll < kL)
      acc += xz[((size_t)(b * kL + ll)) * kDI + d2] * w[k];
  }
  acc = siluf(acc);
  if (d2 < kDH)
    u[(size_t)bl * kDH + d2] = acc;
  else
    ycat[(size_t)bl * kDI + d2] = acc;
}

// ---------------------------------------------------------------------------
// Chunked selective scan. s_t = exp(dt*A)*s_{t-1} + dt*u*B_t is a linear
// recurrence; chunk A-product collapses to exp(A * sum(dt)).
// Phase 1: per (b,d,chunk) compute chunk-local end state (s_in=0) + sum(dt).
// Phase 2: per (b,d) 16-step prefix over chunks -> true initial states.
// Phase 3: per (b,d,chunk) recompute with true s_init, emit y.
// Lanes are consecutive in d: delta/u coalesced, B/C wave-uniform broadcast.
// ---------------------------------------------------------------------------
__device__ __forceinline__ void load_Ac(const float* A_log, int d,
                                        float (&Ac)[kNS]) {
#pragma unroll
  for (int n = 0; n < kNS; ++n) Ac[n] = -__expf(A_log[d * kNS + n]);
}

__launch_bounds__(256)
__global__ void scan_phase1(const float* __restrict__ delta,
                            const float* __restrict__ u,
                            const float* __restrict__ xdbl,
                            const float* __restrict__ A_log,
                            float* __restrict__ s_end,
                            float* __restrict__ sumdt) {
  const int t = blockIdx.x * blockDim.x + threadIdx.x;  // kB*kNC*kDH
  const int d = t & (kDH - 1);
  const int rest = t >> 10;
  const int c = rest & (kNC - 1);
  const int b = rest >> 4;

  float Ac[kNS];
  load_Ac(A_log, d, Ac);
  float s[kNS];
#pragma unroll
  for (int n = 0; n < kNS; ++n) s[n] = 0.0f;
  float sd = 0.0f;

  const int l0 = c * kCL;
  const float* dl = delta + ((size_t)b * kL + l0) * kDH + d;
  const float* ul = u     + ((size_t)b * kL + l0) * kDH + d;
  const float* xb = xdbl  + ((size_t)b * kL + l0) * kXD;

  for (int i = 0; i < kCL; ++i) {
    const float dt = dl[(size_t)i * kDH];
    const float uu = ul[(size_t)i * kDH];
    const float4* bv = reinterpret_cast<const float4*>(xb + (size_t)i * kXD + kRNK);
    float4 B0 = bv[0], B1 = bv[1], B2 = bv[2], B3 = bv[3];
    const float Bv[kNS] = {B0.x,B0.y,B0.z,B0.w, B1.x,B1.y,B1.z,B1.w,
                           B2.x,B2.y,B2.z,B2.w, B3.x,B3.y,B3.z,B3.w};
    const float du = dt * uu;
    sd += dt;
#pragma unroll
    for (int n = 0; n < kNS; ++n)
      s[n] = s[n] * __expf(dt * Ac[n]) + du * Bv[n];
  }
#pragma unroll
  for (int n = 0; n < kNS; ++n)
    s_end[(((size_t)b * kNC + c) * kNS + n) * kDH + d] = s[n];
  sumdt[((size_t)b * kNC + c) * kDH + d] = sd;
}

__launch_bounds__(256)
__global__ void scan_phase2(const float* __restrict__ s_end,
                            const float* __restrict__ sumdt,
                            const float* __restrict__ A_log,
                            float* __restrict__ s_init) {
  const int t = blockIdx.x * blockDim.x + threadIdx.x;  // kB*kDH
  const int d = t & (kDH - 1);
  const int b = t >> 10;

  float Ac[kNS];
  load_Ac(A_log, d, Ac);
  float S[kNS];
#pragma unroll
  for (int n = 0; n < kNS; ++n) S[n] = 0.0f;

  for (int c = 0; c < kNC; ++c) {
    const size_t base = ((size_t)b * kNC + c) * kNS;
#pragma unroll
    for (int n = 0; n < kNS; ++n)
      s_init[(base + n) * kDH + d] = S[n];
    const float sd = sumdt[((size_t)b * kNC + c) * kDH + d];
#pragma unroll
    for (int n = 0; n < kNS; ++n)
      S[n] = S[n] * __expf(sd * Ac[n]) + s_end[(base + n) * kDH + d];
  }
}

__launch_bounds__(256)
__global__ void scan_phase3(const float* __restrict__ delta,
                            const float* __restrict__ u,
                            const float* __restrict__ xdbl,
                            const float* __restrict__ A_log,
                            const float* __restrict__ Dp,
                            const float* __restrict__ s_init,
                            float* __restrict__ ycat) {
  const int t = blockIdx.x * blockDim.x + threadIdx.x;  // kB*kNC*kDH
  const int d = t & (kDH - 1);
  const int rest = t >> 10;
  const int c = rest & (kNC - 1);
  const int b = rest >> 4;

  float Ac[kNS];
  load_Ac(A_log, d, Ac);
  float s[kNS];
#pragma unroll
  for (int n = 0; n < kNS; ++n)
    s[n] = s_init[(((size_t)b * kNC + c) * kNS + n) * kDH + d];
  const float Dd = Dp[d];

  const int l0 = c * kCL;
  const float* dl = delta + ((size_t)b * kL + l0) * kDH + d;
  const float* ul = u     + ((size_t)b * kL + l0) * kDH + d;
  const float* xb = xdbl  + ((size_t)b * kL + l0) * kXD;
  float*       yb = ycat  + ((size_t)b * kL + l0) * kDI + d;

  for (int i = 0; i < kCL; ++i) {
    const float dt = dl[(size_t)i * kDH];
    const float uu = ul[(size_t)i * kDH];
    const float4* bv = reinterpret_cast<const float4*>(xb + (size_t)i * kXD + kRNK);
    float4 B0 = bv[0], B1 = bv[1], B2 = bv[2], B3 = bv[3];
    float4 C0 = bv[4], C1 = bv[5], C2 = bv[6], C3 = bv[7];
    const float Bv[kNS] = {B0.x,B0.y,B0.z,B0.w, B1.x,B1.y,B1.z,B1.w,
                           B2.x,B2.y,B2.z,B2.w, B3.x,B3.y,B3.z,B3.w};
    const float Cv[kNS] = {C0.x,C0.y,C0.z,C0.w, C1.x,C1.y,C1.z,C1.w,
                           C2.x,C2.y,C2.z,C2.w, C3.x,C3.y,C3.z,C3.w};
    const float du = dt * uu;
    float y = 0.0f;
#pragma unroll
    for (int n = 0; n < kNS; ++n) {
      s[n] = s[n] * __expf(dt * Ac[n]) + du * Bv[n];
      y += s[n] * Cv[n];
    }
    yb[(size_t)i * kDI] = y + uu * Dd;
  }
}

// ---------------------------------------------------------------------------
extern "C" void kernel_launch(void* const* d_in, const int* in_sizes, int n_in,
                              void* d_out, int out_size, void* d_ws,
                              size_t ws_size, hipStream_t stream) {
  const float* H     = (const float*)d_in[0];  // (B,L,DM)
  const float* W_in  = (const float*)d_in[1];  // (DI,DM)
  const float* cxw   = (const float*)d_in[2];  // (DH,1,4)
  const float* czw   = (const float*)d_in[3];  // (DH,1,4)
  const float* W_xp  = (const float*)d_in[4];  // (96,DH)
  const float* W_dt  = (const float*)d_in[5];  // (DH,64)
  const float* b_dt  = (const float*)d_in[6];  // (DH,)
  const float* A_log = (const float*)d_in[7];  // (DH,16)
  const float* Dp    = (const float*)d_in[8];  // (DH,)
  const float* W_out = (const float*)d_in[9];  // (DM,DI)
  float* out = (float*)d_out;

  float* ws     = (float*)d_ws;
  float* xz     = ws + OFF_XZ;
  float* u      = ws + OFF_U;
  float* xdbl   = ws + OFF_XDBL;
  float* delta  = ws + OFF_DELTA;
  float* ycat   = ws + OFF_YCAT;
  float* s_end  = ws + OFF_SEND;
  float* s_init = ws + OFF_SINIT;
  float* sumdt  = ws + OFF_SUMDT;

  // 1) in_proj: xz = H * W_in^T   (4096x1024x2048)
  {
    int waves = (kM / 16) * (kDI / 64);
    wmma_gemm_f32bf16<4, 0><<<(waves + 7) / 8, 256, 0, stream>>>(
        H, W_in, xz, nullptr, kM, kDI, kDM, kDM, kDM, kDI);
  }
  // 2) depthwise conv + SiLU
  conv_silu_kernel<<<(kM * kDI) / 256, 256, 0, stream>>>(xz, cxw, czw, u, ycat);
  // 3) x_proj: x_dbl = u * W_xp^T (4096x1024x96)
  {
    int waves = (kM / 16) * (kXD / 32);
    wmma_gemm_f32bf16<2, 0><<<(waves + 7) / 8, 256, 0, stream>>>(
        u, W_xp, xdbl, nullptr, kM, kXD, kDH, kDH, kDH, kXD);
  }
  // 4) dt_proj + fused softplus(acc + 2*b_dt): delta (4096x64x1024)
  {
    int waves = (kM / 16) * (kDH / 64);
    wmma_gemm_f32bf16<4, 1><<<(waves + 7) / 8, 256, 0, stream>>>(
        xdbl, W_dt, delta, b_dt, kM, kDH, kRNK, kXD, kRNK, kDH);
  }
  // 5) chunked selective scan -> ycat[:, 0:1024]
  scan_phase1<<<(kB * kNC * kDH) / 256, 256, 0, stream>>>(delta, u, xdbl,
                                                          A_log, s_end, sumdt);
  scan_phase2<<<(kB * kDH) / 256, 256, 0, stream>>>(s_end, sumdt, A_log,
                                                    s_init);
  scan_phase3<<<(kB * kNC * kDH) / 256, 256, 0, stream>>>(
      delta, u, xdbl, A_log, Dp, s_init, ycat);
  // 6) out_proj: out = ycat * W_out^T (4096x2048x1024)
  {
    int waves = (kM / 16) * (kDM / 64);
    wmma_gemm_f32bf16<4, 0><<<(waves + 7) / 8, 256, 0, stream>>>(
        ycat, W_out, out, nullptr, kM, kDM, kDI, kDI, kDI, kDM);
  }
}
